// Sequence_70136815944250
// MI455X (gfx1250) — matrix operand order repaired and checked by reading, not compiled, not run on hardware
//
#include <hip/hip_runtime.h>
#include <hip/hip_bf16.h>
#include <math.h>

// Problem constants (match reference)
#define BATCH 128
#define TT    1024
#define HH    512
#define VV    256

#define NWG     64      // persistent workgroups: 2 batch-halves x 32 col-groups
#define THREADS 256     // 8 wave32 per WG

typedef __attribute__((ext_vector_type(16))) __bf16 bf16x16;
typedef __attribute__((ext_vector_type(8)))  __bf16 bf16x8;
typedef __attribute__((ext_vector_type(8)))  float  f32x8;

// Branch-free activations (avoid ocml tanhf's divergent range split on the
// serial critical path). Clamp keeps exp finite so (e-1)/(e+1) never hits
// inf/inf; __expf lowers to v_exp_f32.
__device__ __forceinline__ float sigmoidf_(float z) { return 1.0f / (1.0f + __expf(-z)); }
__device__ __forceinline__ float tanhf_(float z) {
  z = fminf(fmaxf(z, -15.0f), 15.0f);
  const float e2 = __expf(2.0f * z);
  return 1.0f - 2.0f / (e2 + 1.0f);
}

__device__ __forceinline__ f32x8 wmma_bf16(bf16x16 a, bf16x16 b, f32x8 c) {
  // v_wmma_f32_16x16x32_bf16  D = A*B + C  (f32 accumulate)
  return __builtin_amdgcn_wmma_f32_16x16x32_bf16(false, a, false, b, (short)0, c, false, false);
}

// A-fragment (16x32 bf16, row-major source, stride HH):
// lanes 0-15 hold K[k0..k0+15]; lanes 16-31 hold K[k0+16..k0+31]
__device__ __forceinline__ bf16x16 load_a_frag(const __bf16* __restrict__ base,
                                               int row, int k0, int lane) {
  const int sel = lane >> 4;
  const __bf16* p = base + row * HH + k0 + sel * 16;
  bf16x8 lo = *(const bf16x8*)p;
  bf16x8 hi = *(const bf16x8*)(p + 8);
  bf16x16 a;
#pragma unroll
  for (int i = 0; i < 8; ++i) { a[i] = lo[i]; a[i + 8] = hi[i]; }
  return a;
}

// Grid-wide barrier: one fresh counter per use (zeroed by hipMemsetAsync each launch).
__device__ __forceinline__ void grid_barrier(unsigned* cnt, unsigned target) {
  __syncthreads();
  if (threadIdx.x == 0) {
    __threadfence();
    __hip_atomic_fetch_add(cnt, 1u, __ATOMIC_RELEASE, __HIP_MEMORY_SCOPE_AGENT);
    while (__hip_atomic_load(cnt, __ATOMIC_ACQUIRE, __HIP_MEMORY_SCOPE_AGENT) < target)
      __builtin_amdgcn_s_sleep(1);   // short poll: barriers are on the critical path
    __threadfence();
  }
  __syncthreads();
}

// ---------------------------------------------------------------------------
// Persistent 2-layer LSTM scan. One grid barrier per timestep.
// WG owns batch rows [mhalf*64, +64) and h-columns [jgrp*16, +16) of BOTH
// layers (all 4 gate quadrants for those columns), so the pointwise LSTM is
// WG-local and the cell state c lives in registers for the whole sequence.
// Dynamic LDS: Whh1 | Wih2 | Whh2 slices (bf16, 64 cols x 512 K each) + gate tile.
// ---------------------------------------------------------------------------
extern "C" __global__ void __launch_bounds__(THREADS, 1)
lstm_persistent(const float* __restrict__ x,
                const float* __restrict__ Wih1, const float* __restrict__ Whh1,
                const float* __restrict__ bih1, const float* __restrict__ bhh1,
                const float* __restrict__ Wih2, const float* __restrict__ Whh2,
                const float* __restrict__ bih2, const float* __restrict__ bhh2,
                __bf16* __restrict__ h1buf,  // [2][BATCH][HH] bf16 (double buffer)
                __bf16* __restrict__ h2buf,  // [2][BATCH][HH] bf16
                __bf16* __restrict__ h2all,  // [TT*BATCH][HH] bf16
                unsigned* __restrict__ counters) {
  extern __shared__ char smem[];
  __bf16* w1  = (__bf16*)smem;        // 64*512 bf16 = 64 KB
  __bf16* w2i = w1 + 64 * HH;         // 64 KB
  __bf16* w2h = w2i + 64 * HH;        // 64 KB
  float*  gts = (float*)(w2h + 64 * HH); // 64 rows x 64 gate cols f32 = 16 KB

  const int tid   = threadIdx.x;
  const int wg    = blockIdx.x;
  const int mhalf = wg & 1;       // which 64-row batch half
  const int jgrp  = wg >> 1;      // which 16-col h slice (0..31)
  const int lane  = tid & 31;
  const int wave  = tid >> 5;
  const int sel   = lane >> 4;
  const int n16l  = lane & 15;
  const int Mt    = wave >> 1;          // 0..3 : 16-row tile within the 64-row half
  const int Nt0   = (wave & 1) * 2;     // this wave does gate quadrants Nt0, Nt0+1

  // ---- stage weight slices into LDS as bf16 (one-time) ----
  for (int idx = tid; idx < 64 * HH; idx += THREADS) {
    const int nl = idx >> 9;            // local gate col = q*16 + n16
    const int k  = idx & (HH - 1);
    const int n  = (nl >> 4) * HH + jgrp * 16 + (nl & 15); // global gate row
    w1[idx]  = (__bf16)Whh1[n * HH + k];
    w2i[idx] = (__bf16)Wih2[n * HH + k];
    w2h[idx] = (__bf16)Whh2[n * HH + k];
  }

  // ---- per-thread pointwise constants + persistent cell state ----
  const int j16  = tid & 15;
  const int bg   = tid >> 4;            // 16 row-groups of 4 rows
  const int jcol = jgrp * 16 + j16;     // global h column
  float wih1r[4], bias1r[4], bias2r[4];
#pragma unroll
  for (int q = 0; q < 4; ++q) {
    const int n = q * HH + jcol;
    wih1r[q]  = Wih1[n];                // IN == 1 -> rank-1 input term
    bias1r[q] = bih1[n] + bhh1[n];
    bias2r[q] = bih2[n] + bhh2[n];
  }
  float c1[4] = {0.f, 0.f, 0.f, 0.f};
  float c2[4] = {0.f, 0.f, 0.f, 0.f};

  // zero this WG's slices of both parities of h1/h2 (t=0 reads parity 1)
#pragma unroll
  for (int pp = 0; pp < 2; ++pp) {
#pragma unroll
    for (int e = 0; e < 4; ++e) {
      const int brow = mhalf * 64 + bg * 4 + e;
      h1buf[pp * BATCH * HH + brow * HH + jcol] = (__bf16)0.0f;
      h2buf[pp * BATCH * HH + brow * HH + jcol] = (__bf16)0.0f;
    }
  }

  grid_barrier(&counters[0], NWG);

  for (int t = 0; t < TT; ++t) {
    const int p = t & 1;
    const __bf16* h1prev = h1buf + (p ^ 1) * BATCH * HH;
    __bf16*       h1cur  = h1buf + p * BATCH * HH;
    const __bf16* h2prev = h2buf + (p ^ 1) * BATCH * HH;
    __bf16*       h2cur  = h2buf + p * BATCH * HH;
    const int arow = mhalf * 64 + Mt * 16 + n16l;

    // ---- layer 1 gate GEMM slice: h1prev @ Whh1^T ----
    {
      f32x8 acc0 = {}, acc1 = {};
#pragma unroll 4
      for (int kk = 0; kk < 16; ++kk) {
        const int k0 = kk * 32;
        bf16x16 a  = load_a_frag(h1prev, arow, k0, lane);
        bf16x16 b0 = *(const bf16x16*)(w1 + (Nt0 * 16 + n16l) * HH + k0 + sel * 16);
        bf16x16 b1 = *(const bf16x16*)(w1 + ((Nt0 + 1) * 16 + n16l) * HH + k0 + sel * 16);
        acc0 = wmma_bf16(a, b0, acc0);
        acc1 = wmma_bf16(a, b1, acc1);
      }
#pragma unroll
      for (int r = 0; r < 8; ++r) {
        const int ml = Mt * 16 + r + sel * 8;
        gts[ml * 64 + Nt0 * 16 + n16l]       = acc0[r];
        gts[ml * 64 + (Nt0 + 1) * 16 + n16l] = acc1[r];
      }
    }
    __syncthreads();

    // ---- pointwise LSTM layer 1 (adds rank-1 x term + biases) ----
#pragma unroll
    for (int e = 0; e < 4; ++e) {
      const int bl   = bg * 4 + e;
      const int brow = mhalf * 64 + bl;
      const float xv = x[brow * TT + t];
      const float gi = gts[bl * 64 +  0 + j16] + xv * wih1r[0] + bias1r[0];
      const float gf = gts[bl * 64 + 16 + j16] + xv * wih1r[1] + bias1r[1];
      const float gg = gts[bl * 64 + 32 + j16] + xv * wih1r[2] + bias1r[2];
      const float go = gts[bl * 64 + 48 + j16] + xv * wih1r[3] + bias1r[3];
      c1[e] = sigmoidf_(gf) * c1[e] + sigmoidf_(gi) * tanhf_(gg);
      h1cur[brow * HH + jcol] = (__bf16)(sigmoidf_(go) * tanhf_(c1[e]));
    }

    grid_barrier(&counters[1 + t], NWG);  // publish h1cur; h2prev already visible

    // ---- layer 2 gate GEMM slice: h1cur @ Wih2^T + h2prev @ Whh2^T ----
    {
      f32x8 acc0 = {}, acc1 = {};
#pragma unroll 2
      for (int kk = 0; kk < 16; ++kk) {
        const int k0 = kk * 32;
        bf16x16 a1  = load_a_frag(h1cur,  arow, k0, lane);
        bf16x16 a2  = load_a_frag(h2prev, arow, k0, lane);
        bf16x16 bi0 = *(const bf16x16*)(w2i + (Nt0 * 16 + n16l) * HH + k0 + sel * 16);
        bf16x16 bh0 = *(const bf16x16*)(w2h + (Nt0 * 16 + n16l) * HH + k0 + sel * 16);
        bf16x16 bi1 = *(const bf16x16*)(w2i + ((Nt0 + 1) * 16 + n16l) * HH + k0 + sel * 16);
        bf16x16 bh1 = *(const bf16x16*)(w2h + ((Nt0 + 1) * 16 + n16l) * HH + k0 + sel * 16);
        acc0 = wmma_bf16(a1, bi0, acc0);
        acc0 = wmma_bf16(a2, bh0, acc0);
        acc1 = wmma_bf16(a1, bi1, acc1);
        acc1 = wmma_bf16(a2, bh1, acc1);
      }
#pragma unroll
      for (int r = 0; r < 8; ++r) {
        const int ml = Mt * 16 + r + sel * 8;
        gts[ml * 64 + Nt0 * 16 + n16l]       = acc0[r];
        gts[ml * 64 + (Nt0 + 1) * 16 + n16l] = acc1[r];
      }
    }
    __syncthreads();

    // ---- pointwise LSTM layer 2; emit h2 (state + per-step history) ----
#pragma unroll
    for (int e = 0; e < 4; ++e) {
      const int bl   = bg * 4 + e;
      const int brow = mhalf * 64 + bl;
      const float gi = gts[bl * 64 +  0 + j16] + bias2r[0];
      const float gf = gts[bl * 64 + 16 + j16] + bias2r[1];
      const float gg = gts[bl * 64 + 32 + j16] + bias2r[2];
      const float go = gts[bl * 64 + 48 + j16] + bias2r[3];
      c2[e] = sigmoidf_(gf) * c2[e] + sigmoidf_(gi) * tanhf_(gg);
      const float hv = sigmoidf_(go) * tanhf_(c2[e]);
      const __bf16 hb = (__bf16)hv;
      h2cur[brow * HH + jcol] = hb;
      h2all[(t * BATCH + brow) * HH + jcol] = hb;
    }
    __syncthreads();  // protect gts against next iteration's stores
  }
}

// ---------------------------------------------------------------------------
// Batched linear + softmax: logits[m=(t*B+b)][v] = h2all[m]·Wlin[v] + blin[v]
// M = T*B = 131072, N = 256, K = 512. Wlin staged bf16 in LDS (256 KB).
// One wave per 16-row M-tile, 16 accumulator N-tiles, in-register softmax.
// ---------------------------------------------------------------------------
extern "C" __global__ void __launch_bounds__(256, 1)
linear_softmax(const __bf16* __restrict__ h2all,
               const float* __restrict__ Wlin,
               const float* __restrict__ blin,
               float* __restrict__ out) {
  extern __shared__ char smem[];
  __bf16* wl = (__bf16*)smem;  // [VV][HH] bf16
  const int tid = threadIdx.x;
  for (int idx = tid; idx < VV * HH; idx += 256) wl[idx] = (__bf16)Wlin[idx];
  __syncthreads();

  const int lane  = tid & 31;
  const int wave  = tid >> 5;
  const int sel   = lane >> 4;
  const int n16l  = lane & 15;
  const int Mbase = blockIdx.x * 128 + wave * 16;
  const int arow  = Mbase + n16l;

  f32x8 acc[16];
#pragma unroll
  for (int nt = 0; nt < 16; ++nt) acc[nt] = (f32x8){};

  for (int kk = 0; kk < 16; ++kk) {
    const int k0 = kk * 32;
    bf16x16 a = load_a_frag(h2all, arow, k0, lane);
#pragma unroll
    for (int nt = 0; nt < 16; ++nt) {
      bf16x16 b = *(const bf16x16*)(wl + (nt * 16 + n16l) * HH + k0 + sel * 16);
      acc[nt] = wmma_bf16(a, b, acc[nt]);
    }
  }

  float bv[16];
#pragma unroll
  for (int nt = 0; nt < 16; ++nt) bv[nt] = blin[nt * 16 + n16l];

#pragma unroll
  for (int r = 0; r < 8; ++r) {
    float mx = -INFINITY;
#pragma unroll
    for (int nt = 0; nt < 16; ++nt) { acc[nt][r] += bv[nt]; mx = fmaxf(mx, acc[nt][r]); }
#pragma unroll
    for (int m = 1; m < 16; m <<= 1) mx = fmaxf(mx, __shfl_xor(mx, m, 32));
    float s = 0.f;
#pragma unroll
    for (int nt = 0; nt < 16; ++nt) { const float e = __expf(acc[nt][r] - mx); acc[nt][r] = e; s += e; }
#pragma unroll
    for (int m = 1; m < 16; m <<= 1) s += __shfl_xor(s, m, 32);
    const float inv = 1.0f / s;
    const int m_  = Mbase + r + sel * 8;
    const int trw = m_ >> 7;       // / BATCH
    const int brw = m_ & 127;      // % BATCH
    float* orow = out + (size_t)brw * TT * VV + (size_t)trw * VV;
#pragma unroll
    for (int nt = 0; nt < 16; ++nt) orow[nt * 16 + n16l] = acc[nt][r] * inv;
  }
}

// ---------------------------------------------------------------------------
extern "C" void kernel_launch(void* const* d_in, const int* in_sizes, int n_in,
                              void* d_out, int out_size, void* d_ws, size_t ws_size,
                              hipStream_t stream) {
  (void)in_sizes; (void)n_in; (void)out_size; (void)ws_size;
  const float* x    = (const float*)d_in[0];
  const float* Wih1 = (const float*)d_in[1];
  const float* Whh1 = (const float*)d_in[2];
  const float* bih1 = (const float*)d_in[3];
  const float* bhh1 = (const float*)d_in[4];
  const float* Wih2 = (const float*)d_in[5];
  const float* Whh2 = (const float*)d_in[6];
  const float* bih2 = (const float*)d_in[7];
  const float* bhh2 = (const float*)d_in[8];
  const float* Wlin = (const float*)d_in[9];
  const float* blin = (const float*)d_in[10];

  // ws layout: counters (8 KB) | h1[2]x128x512 bf16 | h2[2] | h2_all[T*B*H] bf16 (~134.8 MB)
  unsigned* counters = (unsigned*)d_ws;
  __bf16* h1buf = (__bf16*)((char*)d_ws + 8192);
  __bf16* h2buf = h1buf + 2 * BATCH * HH;
  __bf16* h2all = h2buf + 2 * BATCH * HH;

  hipMemsetAsync(counters, 0, (TT + 2) * sizeof(unsigned), stream);

  const size_t lds1 = (size_t)(3 * 64 * HH) * sizeof(__bf16) + (size_t)(64 * 64) * sizeof(float); // 208 KB
  lstm_persistent<<<NWG, THREADS, lds1, stream>>>(x, Wih1, Whh1, bih1, bhh1,
                                                  Wih2, Whh2, bih2, bhh2,
                                                  h1buf, h2buf, h2all, counters);

  const size_t lds2 = (size_t)VV * HH * sizeof(__bf16); // 256 KB
  linear_softmax<<<(TT * BATCH) / 128, 256, lds2, stream>>>(h2all, Wlin, blin, (float*)d_out);
}
